// AttnShaper_8452495639099
// MI455X (gfx1250) — matrix operand order repaired and checked by reading, not compiled, not run on hardware
//
#include <hip/hip_runtime.h>
#include <hip/hip_bf16.h>

// ---------------------------------------------------------------------------
// AttnShaper for MI455X (gfx1250, wave32, WMMA bf16 16x16x32)
//   B=256, L=65536, HIST=256, WIN=64, DIM=256, NPAT=64, T=L/WIN=1024
// ---------------------------------------------------------------------------

#define BATCH 256
#define LSEQ  65536
#define HIST  256
#define WIN   64
#define DIM   256
#define NPAT  64
#define TSTEPS 1024

typedef __attribute__((ext_vector_type(16))) __bf16 v16bf;
typedef __attribute__((ext_vector_type(8)))  __bf16 v8bf;
typedef __attribute__((ext_vector_type(8)))  float  v8f;
typedef __attribute__((ext_vector_type(4)))  int    v4i;

// ---- CDNA5 async global->LDS path (guarded; sync fallback otherwise) ------
#if defined(__gfx1250__) && __has_builtin(__builtin_amdgcn_global_load_async_to_lds_b128)
#define ASYNC_LDS 1
#else
#define ASYNC_LDS 0
#endif

// Move 16 bytes (8 bf16) global -> LDS.
// Builtin signature (from hipcc diagnostic): (v4i* src, v4i* lds, imm offset, imm cpol)
__device__ __forceinline__ void copy16B_to_lds(const __bf16* g, __bf16* l) {
#if ASYNC_LDS
  __builtin_amdgcn_global_load_async_to_lds_b128((v4i*)g, (v4i*)l, 0, 0);
#else
  *(v8bf*)l = *(const v8bf*)g;
#endif
}
__device__ __forceinline__ void async_fence() {
#if ASYNC_LDS
#if __has_builtin(__builtin_amdgcn_s_wait_asynccnt)
  __builtin_amdgcn_s_wait_asynccnt(0);
#else
  asm volatile("s_wait_asynccnt 0" ::: "memory");
#endif
#endif
}

// ---------------------------------------------------------------------------
// Kernel 0: precision prep.  conv_w (DIM x HIST row-major == [d][k], exactly
// the B-fragment layout) -> bf16.  keys (DIM x NPAT) -> transposed bf16
// keys_t (NPAT x DIM) so GEMM2 B-fragments are contiguous-in-K.
// ---------------------------------------------------------------------------
__global__ void prep_kernel(const float* __restrict__ conv_w,
                            const float* __restrict__ keys,
                            __bf16* __restrict__ w_bf,
                            __bf16* __restrict__ keys_t) {
  int i = blockIdx.x * 256 + threadIdx.x;
  if (i < DIM * HIST) w_bf[i] = (__bf16)conv_w[i];
  if (i < DIM * NPAT) {
    int d = i >> 6, p = i & 63;           // keys is [d][p]
    keys_t[p * DIM + d] = (__bf16)keys[i];
  }
}

// A-fragment (16-bit A 16x32):
//   lane l: m = l&15 ; element j: k = kk*32 + 16*(l>>4) + j
//   -> 16 contiguous bf16 (32B) along k.
__device__ __forceinline__ v16bf load_frag_a(const __bf16* p) {
  v8bf lo = *(const v8bf*)p;
  v8bf hi = *(const v8bf*)(p + 8);
  return __builtin_shufflevector(lo, hi, 0,1,2,3,4,5,6,7,8,9,10,11,12,13,14,15);
}
// B-fragment (16-bit B 32x16): lane l: n = l&15 ; element j: k = (l>>4)*16 + j
//   -> 16 contiguous bf16 (32B) along k.
__device__ __forceinline__ v16bf load_frag_b(const __bf16* p) {
  v8bf lo = *(const v8bf*)p;
  v8bf hi = *(const v8bf*)(p + 8);
  return __builtin_shufflevector(lo, hi, 0,1,2,3,4,5,6,7,8,9,10,11,12,13,14,15);
}

// ---------------------------------------------------------------------------
// Kernel 1: fused conv(im2col GEMM1) + bias + ReLU, then GEMM2 vs keys,
// relu6 -> scores[t][b][p].  One workgroup = one batch row x 64 time steps.
//
// LDS plan (117 KB, 2 WGs/WGP within the 320 KB pool):
//   x_lds    : window rows alias with stride 64 (im2col is free)      8.5 KB
//   out_lds  : relu(conv) 64t x 256d bf16, stride 264 (pad)          33 KB
//   wslab[2] : double-buffered weight K-slab [256d][40] (stride-40
//              padding -> conflict-free 16B ds_load_b128 frags)      40 KB
//   keys_lds : keys^T [64p][264]                                     33 KB
// Weights/keys are staged ONCE per WG (128+32 KB of L2 traffic instead of
// 512+128 KB of redundant per-wave global fragment reads).
// ---------------------------------------------------------------------------
__global__ __launch_bounds__(256)
void conv_score_kernel(const float* __restrict__ x,
                       const float* __restrict__ conv_b,
                       const __bf16* __restrict__ w_bf,
                       const __bf16* __restrict__ keys_t,
                       float* __restrict__ scores) {
  __shared__ __bf16 x_lds[4352];
  __shared__ __bf16 out_lds[64 * 264];
  __shared__ __bf16 wslab[2][256 * 40];
  __shared__ __bf16 keys_lds[64 * 264];

  const int b   = blockIdx.y;
  const int t0  = blockIdx.x * 64;          // 64 time steps per WG
  const int tid = threadIdx.x;
  const int wave = tid >> 5;                // 8 waves
  const int l    = tid & 31;
  const int lo16 = l & 15;
  const int hi   = l >> 4;                  // 0/1

  // ---- stage keys^T (once) and weight slab 0 (async if available) ----
  #pragma unroll
  for (int c = tid; c < 2048; c += 256) {   // 64 rows x 32 chunks of 16B
    int p = c >> 5, seg = c & 31;
    copy16B_to_lds(keys_t + p * DIM + seg * 8, &keys_lds[p * 264 + seg * 8]);
  }
  {
    const __bf16* src = w_bf + tid * HIST;  // d = tid, k = 0..15
    copy16B_to_lds(src,     &wslab[0][tid * 40]);
    copy16B_to_lds(src + 8, &wslab[0][tid * 40 + 8]);
  }

  // ---- stage x span [t0*64-255 .. t0*64+4032] as bf16, left zero-pad ----
  {
    const int base = t0 * WIN - (HIST - 1);
    const float* xb = x + b * LSEQ;
    for (int i = tid; i < 4288; i += 256) {
      int gi = base + i;
      float v = (gi >= 0) ? xb[gi] : 0.0f;
      x_lds[i] = (__bf16)v;
    }
  }
  async_fence();
  __syncthreads();

  // ---- GEMM1: (64 x 256) = im2col(x) @ w^T ; wave -> (M-tile, dim-half) ----
  {
    const int mt = wave & 3;                // M tile: t rows mt*16..+15
    const int nh = wave >> 2;               // dim half: 0 -> d<128, 1 -> d>=128
    const int mrow = mt * 16 + lo16;

    v8f acc[8];
    #pragma unroll
    for (int n = 0; n < 8; ++n) { v8f z = {}; acc[n] = z; }

    #pragma unroll
    for (int kk = 0; kk < 8; ++kk) {        // K = 256 = 8 x 32
      // stage next K-slab into the other buffer while computing this one
      if (kk < 7) {
        const __bf16* src = w_bf + tid * HIST + (kk + 1) * 32;
        __bf16* dst = &wslab[(kk + 1) & 1][tid * 40];
        copy16B_to_lds(src,     dst);
        copy16B_to_lds(src + 8, dst + 8);
      }
      const __bf16* wb = wslab[kk & 1];
      v16bf a = load_frag_a(&x_lds[mrow * WIN + kk * 32 + hi * 16]);
      #pragma unroll
      for (int nt = 0; nt < 8; ++nt) {
        const int d = nh * 128 + nt * 16 + lo16;
        v16bf bb = load_frag_b(wb + d * 40 + hi * 16);
        acc[nt] = __builtin_amdgcn_wmma_f32_16x16x32_bf16(
            false, a, false, bb, (short)0, acc[nt], false, false);
      }
      if (kk < 7) { async_fence(); __syncthreads(); }
    }

    // epilogue: relu(acc + bias) -> out_lds (bf16); D layout: m=r+8*hi, n=l&15
    #pragma unroll
    for (int nt = 0; nt < 8; ++nt) {
      const int d = nh * 128 + nt * 16 + lo16;
      const float bias = conv_b[d];
      #pragma unroll
      for (int r = 0; r < 8; ++r) {
        float v = fmaxf(acc[nt][r] + bias, 0.0f);
        int m = mt * 16 + r + 8 * hi;
        out_lds[m * 264 + d] = (__bf16)v;
      }
    }
  }
  __syncthreads();

  // ---- GEMM2: scores = relu6(out @ keys) : M=64 (4 tiles) x N=64 (4 tiles) --
  {
    const int mt = wave & 3;
    const int nh = wave >> 2;               // this wave: p-tiles nh and nh+2
    const int mrow = mt * 16 + lo16;

    v8f acc0 = {}, acc1 = {};
    #pragma unroll
    for (int kk = 0; kk < 8; ++kk) {
      v16bf a = load_frag_a(&out_lds[mrow * 264 + kk * 32 + hi * 16]);
      {
        const int p = nh * 16 + lo16;
        v16bf bb = load_frag_b(&keys_lds[p * 264 + kk * 32 + hi * 16]);
        acc0 = __builtin_amdgcn_wmma_f32_16x16x32_bf16(
            false, a, false, bb, (short)0, acc0, false, false);
      }
      {
        const int p = (nh + 2) * 16 + lo16;
        v16bf bb = load_frag_b(&keys_lds[p * 264 + kk * 32 + hi * 16]);
        acc1 = __builtin_amdgcn_wmma_f32_16x16x32_bf16(
            false, a, false, bb, (short)0, acc1, false, false);
      }
    }

    #pragma unroll
    for (int h = 0; h < 2; ++h) {
      const int p = (nh + 2 * h) * 16 + lo16;
      #pragma unroll
      for (int r = 0; r < 8; ++r) {
        float v = h ? acc1[r] : acc0[r];
        v = fminf(fmaxf(v, 0.0f), 6.0f);
        int t = t0 + mt * 16 + r + 8 * hi;
        scores[t * (BATCH * NPAT) + b * NPAT + p] = v;
      }
    }
  }
}

// ---------------------------------------------------------------------------
// Kernel 2: sequential softmax scan over T.  One wave per block; each wave
// carries FOUR independent batch chains so the shuffle/exp dependency chains
// interleave (latency hiding via ILP).  Lane l owns p = {l, l+32}.
// ---------------------------------------------------------------------------
__global__ __launch_bounds__(32)
void scan_kernel(const float* __restrict__ scores,
                 const float* __restrict__ avg0,
                 float* __restrict__ probs) {
  const int l = threadIdx.x;
  const int bbase = blockIdx.x * 4;
  const float inv_p = 1.0f / (float)NPAT;

  float a0[4], a1[4];
  #pragma unroll
  for (int j = 0; j < 4; ++j) {
    a0[j] = avg0[(bbase + j) * NPAT + l];
    a1[j] = avg0[(bbase + j) * NPAT + 32 + l];
  }

  for (int t = 0; t < TSTEPS; ++t) {
    float s0[4], s1[4];
    #pragma unroll
    for (int j = 0; j < 4; ++j) {
      const int base = t * (BATCH * NPAT) + (bbase + j) * NPAT;
      s0[j] = scores[base + l];
      s1[j] = scores[base + 32 + l];
    }
    #pragma unroll
    for (int j = 0; j < 4; ++j) {
      float d0 = s0[j] - a0[j];
      float d1 = s1[j] - a1[j];
      float m = fmaxf(d0, d1);
      #pragma unroll
      for (int off = 16; off > 0; off >>= 1) m = fmaxf(m, __shfl_xor(m, off, 32));
      float e0 = __expf(d0 - m);
      float e1 = __expf(d1 - m);
      float s = e0 + e1;
      #pragma unroll
      for (int off = 16; off > 0; off >>= 1) s += __shfl_xor(s, off, 32);
      float r = 1.0f / s;
      float p0 = e0 * r;
      float p1 = e1 * r;
      a0[j] += p0 - inv_p;
      a1[j] += p1 - inv_p;
      const int ob = (bbase + j) * (TSTEPS * NPAT) + t * NPAT;
      probs[ob + l]      = p0;
      probs[ob + 32 + l] = p1;
    }
  }
}

// ---------------------------------------------------------------------------
// Kernel 3: signal = probs @ shapes, out = relu(signal - x).  Memory bound;
// shapes (16KB) + a 16-t probs tile staged in LDS, VALU dot products.
// ---------------------------------------------------------------------------
__global__ __launch_bounds__(256)
void signal_kernel(const float* __restrict__ probs,
                   const float* __restrict__ shapes,
                   const float* __restrict__ x,
                   float* __restrict__ out) {
  __shared__ float sh[NPAT * WIN];          // shapes [p][w]  (16 KB)
  __shared__ float pr[16][NPAT];            // probs tile     (4 KB)

  const int b  = blockIdx.y;
  const int t0 = blockIdx.x * 16;
  const int tid = threadIdx.x;

  #pragma unroll
  for (int i = tid; i < NPAT * WIN; i += 256) sh[i] = shapes[i];
  #pragma unroll
  for (int i = tid; i < 16 * NPAT; i += 256) {
    int tl = i >> 6, p = i & 63;
    pr[tl][p] = probs[b * (TSTEPS * NPAT) + (t0 + tl) * NPAT + p];
  }
  __syncthreads();

  const int w  = tid & 63;
  const int tb = tid >> 6;                  // 0..3
  #pragma unroll
  for (int i = 0; i < 4; ++i) {
    const int tl = tb + i * 4;
    float acc = 0.0f;
    #pragma unroll
    for (int p = 0; p < NPAT; ++p) acc = fmaf(pr[tl][p], sh[p * WIN + w], acc);
    const int gi = b * LSEQ + (t0 + tl) * WIN + w;
    out[gi] = fmaxf(acc - x[gi], 0.0f);
  }
}

// ---------------------------------------------------------------------------
// Launch
// ---------------------------------------------------------------------------
extern "C" void kernel_launch(void* const* d_in, const int* in_sizes, int n_in,
                              void* d_out, int out_size, void* d_ws, size_t ws_size,
                              hipStream_t stream) {
  const float* x      = (const float*)d_in[0];   // (B, L)
  const float* avg0   = (const float*)d_in[1];   // (B, NPAT)
  const float* conv_w = (const float*)d_in[2];   // (DIM, 1, HIST)
  const float* conv_b = (const float*)d_in[3];   // (DIM,)
  const float* keys   = (const float*)d_in[4];   // (DIM, NPAT)
  const float* shapes = (const float*)d_in[5];   // (NPAT, WIN)
  float* out = (float*)d_out;

  char* ws = (char*)d_ws;
  __bf16* w_bf   = (__bf16*)(ws);                            // 128 KB
  __bf16* keys_t = (__bf16*)(ws + 131072);                   // 32 KB
  float*  scores = (float*)(ws + 163840);                    // 64 MB (T,B,P)
  float*  probs  = (float*)(ws + 163840 + (size_t)TSTEPS * BATCH * NPAT * 4); // 64 MB (B,T,P)

  prep_kernel<<<256, 256, 0, stream>>>(conv_w, keys, w_bf, keys_t);
  conv_score_kernel<<<dim3(TSTEPS / 64, BATCH), 256, 0, stream>>>(
      x, conv_b, w_bf, keys_t, scores);
  scan_kernel<<<BATCH / 4, 32, 0, stream>>>(scores, avg0, probs);
  signal_kernel<<<dim3(TSTEPS / 16, BATCH), 256, 0, stream>>>(probs, shapes, x, out);
}
